// Deepv2_voxelnumtest_74741020885607
// MI455X (gfx1250) — matrix operand order repaired and checked by reading, not compiled, not run on hardware
//
#include <hip/hip_runtime.h>
#include <hip/hip_bf16.h>
#include <cstdint>
#include <cstddef>

typedef __attribute__((ext_vector_type(16))) _Float16 v16h;
typedef __attribute__((ext_vector_type(8)))  _Float16 v8h;
typedef __attribute__((ext_vector_type(8)))  float    v8f;
typedef int v4i __attribute__((vector_size(16)));
typedef __attribute__((address_space(1))) v4i* gptr_v4i;
typedef __attribute__((address_space(3))) v4i* lptr_v4i;

#define BN_EPS 1e-5f

#if __has_builtin(__builtin_amdgcn_global_load_async_to_lds_b128)
#define HAVE_ASYNC_LDS 1
#endif

static __device__ __forceinline__ void async_wait0() {
#if defined(HAVE_ASYNC_LDS)
#if __has_builtin(__builtin_amdgcn_s_wait_asynccnt)
  __builtin_amdgcn_s_wait_asynccnt(0);
#else
  asm volatile("s_wait_asynccnt 0x0" ::: "memory");
#endif
#endif
}

// ---------------------------------------------------------------------------
// Mask handling: bool mask -> float level 0, then 7 stride-2 max-pool levels
// (all grid dims are powers of two -> shift/mask coords, no int division)
// ---------------------------------------------------------------------------
__global__ void k_mask_to_float(const unsigned char* __restrict__ m,
                                float* __restrict__ mf, int n) {
  int i = blockIdx.x * blockDim.x + threadIdx.x;
  if (i < n) mf[i] = m[i] ? 1.0f : 0.0f;
}

__global__ void k_down_mask(const float* __restrict__ mi,
                            float* __restrict__ mo, int Din, int lgDo) {
  int Do = 1 << lgDo;
  int n = Do * Do * Do;
  int i = blockIdx.x * blockDim.x + threadIdx.x;
  if (i >= n) return;
  int ox = i & (Do - 1);
  int oy = (i >> lgDo) & (Do - 1);
  int oz = i >> (2 * lgDo);
  float mv = 0.0f;
  for (int dz = 0; dz < 3; ++dz) {
    int iz = 2 * oz + dz - 1; if ((unsigned)iz >= (unsigned)Din) continue;
    for (int dy = 0; dy < 3; ++dy) {
      int iy = 2 * oy + dy - 1; if ((unsigned)iy >= (unsigned)Din) continue;
      for (int dx = 0; dx < 3; ++dx) {
        int ix = 2 * ox + dx - 1; if ((unsigned)ix >= (unsigned)Din) continue;
        mv = fmaxf(mv, mi[((size_t)iz * Din + iy) * Din + ix]);
      }
    }
  }
  mo[i] = mv;
}

// ---------------------------------------------------------------------------
// Input repack: NCDHW fp32 (already mask-multiplied by setup) -> [site, C] f16
// ---------------------------------------------------------------------------
__global__ void k_pack_input(const float* __restrict__ x,
                             _Float16* __restrict__ act, int nsite, int cin) {
  int i = blockIdx.x * blockDim.x + threadIdx.x;
  if (i >= nsite * cin) return;
  int c = i % cin; int s = i / cin;
  act[(size_t)s * cin + c] = (_Float16)x[(size_t)c * nsite + s];
}

// ---------------------------------------------------------------------------
// BN prefold: inv = gamma * rsqrt(var+eps), off = beta + (bias - mean) * inv
// ---------------------------------------------------------------------------
__global__ void k_bn_prep(const float* __restrict__ g, const float* __restrict__ b,
                          const float* __restrict__ mean, const float* __restrict__ var,
                          const float* __restrict__ bias,
                          float* __restrict__ inv, float* __restrict__ off, int cout) {
  int i = blockIdx.x * blockDim.x + threadIdx.x;
  if (i >= cout) return;
  float iv = g[i] * rsqrtf(var[i] + BN_EPS);
  float o = b[i] - mean[i] * iv;
  if (bias) o += bias[i] * iv;
  inv[i] = iv; off[i] = o;
}

// ---------------------------------------------------------------------------
// Weight prepack: W(cout,cin,3,3,3) fp32 -> B fragments, f16
// layout: [tap][ntPair][kc][t(2)][lane(32)][h(16)]  (contiguous per tap+pair
// for async LDS staging); element = W[n][k][tap]
//   n = (ntPair*2+t)*16 + (lane&15), k = kc*32 + 8*(lane>>4) + (h<8 ? h : h+8)
// ---------------------------------------------------------------------------
__global__ void k_pack_w(const float* __restrict__ w, _Float16* __restrict__ wp,
                         int cin, int cout, int KC, int NPAIR, int total) {
  int i = blockIdx.x * blockDim.x + threadIdx.x;
  if (i >= total) return;
  int h = i & 15;
  int lane = (i >> 4) & 31;
  int rest = i >> 9;
  int t = rest & 1; rest >>= 1;
  int kc = rest % KC; rest /= KC;
  int np = rest % NPAIR;
  int tap = rest / NPAIR;
  int k = kc * 32 + 8 * (lane >> 4) + (h < 8 ? h : h + 8);
  int n = (np * 2 + t) * 16 + (lane & 15);
  float v = 0.0f;
  if (k < cin && n < cout) v = w[((size_t)n * cin + k) * 27 + tap];
  wp[i] = (_Float16)v;
}

// ---------------------------------------------------------------------------
// Implicit-GEMM conv3d (pad 1, stride st) + BN + ReLU + mask, via WMMA f16.
// One wave -> 16 sites x 32 couts (two independent WMMA accumulator chains),
// 4 waves/block on consecutive site tiles sharing one cout pair.
// Deep layers (KC>=2): B fragments async-staged to LDS per tap by wave 0
// (GLOBAL_LOAD_ASYNC_TO_LDS_B128 + s_wait_asynccnt), read back via ds_load.
// ---------------------------------------------------------------------------
__global__ __launch_bounds__(128) void k_conv_wmma(
    const _Float16* __restrict__ x, const _Float16* __restrict__ wp,
    const float* __restrict__ inv, const float* __restrict__ off,
    const float* __restrict__ mout, _Float16* __restrict__ y,
    int cin, int cout, int KC, int NPAIR, int Din, int lgD, int st, int nsite) {
  extern __shared__ _Float16 smem[];
  const int lane = threadIdx.x & 31;
  const int wave = threadIdx.x >> 5;
  const int tile = blockIdx.x * 4 + wave;
  const int np = blockIdx.y;
  const int tileBase = tile * 16;
  if (tileBase >= nsite) return;  // wave-uniform; terminated waves leave barriers

  const int Dout = 1 << lgD;
  const int s = tileBase + (lane & 15);
  const bool srow = s < nsite;
  const int ox = s & (Dout - 1);
  const int oy = (s >> lgD) & (Dout - 1);
  const int oz = s >> (2 * lgD);
  const int koff = (lane >> 4) * 8;
  const bool fullvec = ((cin & 31) == 0);
  const bool useLds = (KC >= 2);

  v8f acc[2] = {{}, {}};
  for (int tap = 0; tap < 27; ++tap) {
    const int dz = tap / 9, dy = (tap / 3) % 3, dx = tap % 3;
    const int iz = oz * st + dz - 1;
    const int iy = oy * st + dy - 1;
    const int ix = ox * st + dx - 1;
    const bool valid = srow && (unsigned)iz < (unsigned)Din &&
                       (unsigned)iy < (unsigned)Din && (unsigned)ix < (unsigned)Din;
    const _Float16* row = x + (size_t)(((size_t)iz * Din + iy) * Din + ix) * cin;
    const _Float16* wTap = wp + ((size_t)tap * NPAIR + np) * (size_t)KC * 1024;

    if (useLds) {
      __syncthreads();           // previous tap's LDS reads complete
      if (wave == 0) {
        const int chunks = KC * 4;           // KC*128 16B chunks / 32 lanes
        for (int j = 0; j < chunks; ++j) {
          const int c16 = j * 32 + lane;
#if defined(HAVE_ASYNC_LDS)
          __builtin_amdgcn_global_load_async_to_lds_b128(
              (gptr_v4i)(wTap + (size_t)c16 * 8),
              (lptr_v4i)(&smem[(size_t)c16 * 8]),
              0, 0);
#else
          *(v8h*)&smem[(size_t)c16 * 8] = *(const v8h*)(wTap + (size_t)c16 * 8);
#endif
        }
        async_wait0();
      }
      __syncthreads();           // staged B visible to all waves
    }

    for (int kc = 0; kc < KC; ++kc) {
      v16h a;
      if (valid && fullvec) {
        // two aligned 16B loads: K {base..base+7} and {base+16..base+23}
        const _Float16* p = row + kc * 32 + koff;
        v8h lo = *(const v8h*)p;
        v8h hi = *(const v8h*)(p + 16);
#pragma unroll
        for (int q = 0; q < 8; ++q) { a[q] = lo[q]; a[8 + q] = hi[q]; }
      } else {
#pragma unroll
        for (int h = 0; h < 16; ++h) {
          int k = kc * 32 + koff + (h < 8 ? h : h + 8);
          a[h] = (valid && k < cin) ? row[k] : (_Float16)0.0f;
        }
      }
      v16h b0, b1;
      if (useLds) {
        b0 = *(const v16h*)(smem + (size_t)(kc * 2 + 0) * 512 + lane * 16);
        b1 = *(const v16h*)(smem + (size_t)(kc * 2 + 1) * 512 + lane * 16);
      } else {
        b0 = *(const v16h*)(wTap + (size_t)(kc * 2 + 0) * 512 + lane * 16);
        b1 = *(const v16h*)(wTap + (size_t)(kc * 2 + 1) * 512 + lane * 16);
        if (kc + 1 < KC)
          __builtin_prefetch((const void*)(wTap + (size_t)(kc + 1) * 1024), 0, 1);
      }
      acc[0] = __builtin_amdgcn_wmma_f32_16x16x32_f16(
          false, a, false, b0, (short)0, acc[0], false, false);
      acc[1] = __builtin_amdgcn_wmma_f32_16x16x32_f16(
          false, a, false, b1, (short)0, acc[1], false, false);
    }
  }

  // Epilogue: D layout lane -> N = lane&15; VGPR r -> M = r + 8*(lane>>4)
  const int nb = lane & 15;
  const int mh = 8 * (lane >> 4);
#pragma unroll
  for (int t = 0; t < 2; ++t) {
    const int n = (np * 2 + t) * 16 + nb;
    if (n < cout) {
      const float iv = inv[n], of = off[n];
#pragma unroll
      for (int r = 0; r < 8; ++r) {
        const int sm = tileBase + r + mh;
        if (sm < nsite) {
          float v = acc[t][r] * iv + of;
          v = v > 0.0f ? v : 0.0f;
          v *= mout[sm];
          y[(size_t)sm * cout + n] = (_Float16)v;
        }
      }
    }
  }
}

// ---------------------------------------------------------------------------
// Finalize one tower: voxel count (int bits) + 1024-ch fp32 feature vector
// ---------------------------------------------------------------------------
__global__ void k_finalize(const float* __restrict__ mfin,
                           const _Float16* __restrict__ act,
                           int* __restrict__ cnt, float* __restrict__ vec, int C) {
  int i = blockIdx.x * blockDim.x + threadIdx.x;
  if (i == 0) *cnt = (int)(mfin[0] + 0.5f);
  if (i < C) vec[i] = (float)act[i];
}

// ---------------------------------------------------------------------------
// Host orchestration
// ---------------------------------------------------------------------------
struct LayerCfg { int cin, cout, st, hasb; };
static const LayerCfg CFGH[20] = {
  {3,16,1,0},{16,24,2,0},{24,24,1,0},{24,32,2,0},{32,32,1,0},{32,64,2,0},
  {64,64,1,0},{64,128,2,0},{128,128,1,0},{128,256,2,0},{256,256,1,0},
  {256,256,1,0},{256,512,2,0},{512,512,1,0},{512,512,1,0},{512,512,1,0},
  {512,512,1,0},{512,1024,2,1},{1024,1024,1,0},{1024,1024,1,0}
};

static inline size_t align256(size_t x) { return (x + 255) & ~(size_t)255; }
static inline int ilog2(int v) { int l = 0; while ((1 << l) < v) ++l; return l; }

extern "C" void kernel_launch(void* const* d_in, const int* in_sizes, int n_in,
                              void* d_out, int out_size, void* d_ws, size_t ws_size,
                              hipStream_t stream) {
  (void)in_sizes; (void)n_in; (void)out_size; (void)ws_size;
  const int G = 128;
  const int G3 = G * G * G;

  // --- workspace carve ---
  char* ws = (char*)d_ws;
  size_t o = 0;
  size_t actBytes = (size_t)G3 * 16 * sizeof(_Float16);           // 67 MB (layer-0 out)
  _Float16* actA = (_Float16*)(ws + o); o += align256(actBytes);
  _Float16* actB = (_Float16*)(ws + o); o += align256(actBytes);
  size_t wpackBytes = (size_t)27 * 32 * 32 * 1024 * sizeof(_Float16);  // 56.6 MB max layer
  _Float16* wpack = (_Float16*)(ws + o); o += align256(wpackBytes);
  float* bnInv = (float*)(ws + o); o += align256(1024 * sizeof(float));
  float* bnOff = (float*)(ws + o); o += align256(1024 * sizeof(float));
  float* mlev[8];
  { int d = G;
    for (int lv = 0; lv < 8; ++lv) {
      mlev[lv] = (float*)(ws + o);
      o += align256((size_t)d * d * d * sizeof(float));
      d >>= 1;
    } }

  // --- parse params (dict order: w, gamma, beta, mean, var, [bias]) ---
  const float *Pw[2][20], *Pg[2][20], *Pb[2][20], *Pm[2][20], *Pv[2][20], *Pbias[2][20];
  int pi = 4;
  for (int net = 0; net < 2; ++net)
    for (int l = 0; l < 20; ++l) {
      Pw[net][l] = (const float*)d_in[pi++];
      Pg[net][l] = (const float*)d_in[pi++];
      Pb[net][l] = (const float*)d_in[pi++];
      Pm[net][l] = (const float*)d_in[pi++];
      Pv[net][l] = (const float*)d_in[pi++];
      Pbias[net][l] = CFGH[l].hasb ? (const float*)d_in[pi++] : nullptr;
    }

  for (int net = 0; net < 2; ++net) {
    const float* xin = (const float*)d_in[net == 0 ? 0 : 2];
    const unsigned char* minb = (const unsigned char*)d_in[net == 0 ? 1 : 3];

    // mask pyramid
    k_mask_to_float<<<(G3 + 255) / 256, 256, 0, stream>>>(minb, mlev[0], G3);
    { int d = G;
      for (int lv = 0; lv < 7; ++lv) {
        int dn = d >> 1; int nOut = dn * dn * dn;
        k_down_mask<<<(nOut + 255) / 256, 256, 0, stream>>>(mlev[lv], mlev[lv + 1],
                                                            d, ilog2(dn));
        d = dn;
      } }

    // input repack
    k_pack_input<<<((size_t)G3 * 3 + 255) / 256, 256, 0, stream>>>(xin, actA, G3, 3);

    _Float16* cur = actA; _Float16* nxt = actB;
    int Din = G, mlvl = 0;
    for (int l = 0; l < 20; ++l) {
      int cin = CFGH[l].cin, cout = CFGH[l].cout, st = CFGH[l].st;
      int Dout = (st == 2) ? (Din >> 1) : Din;
      int nsite = Dout * Dout * Dout;
      int KC = (cin + 31) / 32;
      int NT = (cout + 15) / 16;
      int NPAIR = (NT + 1) / 2;
      int tot = 27 * NPAIR * KC * 1024;

      k_pack_w<<<(tot + 255) / 256, 256, 0, stream>>>(Pw[net][l], wpack,
                                                      cin, cout, KC, NPAIR, tot);
      k_bn_prep<<<(cout + 255) / 256, 256, 0, stream>>>(
          Pg[net][l], Pb[net][l], Pm[net][l], Pv[net][l], Pbias[net][l],
          bnInv, bnOff, cout);

      if (st == 2) mlvl++;
      int tiles = (nsite + 15) / 16;
      dim3 grid((tiles + 3) / 4, NPAIR);
      size_t shBytes = (KC >= 2) ? (size_t)KC * 2048 : 0;   // KC*2*512 halves
      k_conv_wmma<<<grid, 128, shBytes, stream>>>(cur, wpack, bnInv, bnOff,
                                                  mlev[mlvl], nxt,
                                                  cin, cout, KC, NPAIR,
                                                  Din, ilog2(Dout), st, nsite);
      _Float16* t = cur; cur = nxt; nxt = t;
      Din = Dout;
    }

    // outputs: [n_sgm, n_velo, sgm_x(1024), velo_x(1024)]
    int* cnt = ((int*)d_out) + net;
    float* vec = ((float*)d_out) + 2 + net * 1024;
    k_finalize<<<(1024 + 255) / 256, 256, 0, stream>>>(mlev[7], cur, cnt, vec, 1024);
  }
}
